// Laplacian_41540923687075
// MI455X (gfx1250) — matrix operand order repaired and checked by reading, not compiled, not run on hardware
//
#include <hip/hip_runtime.h>

// D=1024, H1=4096, H2=4096, OUT=1
#define DC   1024
#define H1C  4096
#define H2C  4096

typedef __attribute__((ext_vector_type(16))) __bf16 v16bf;
typedef __attribute__((ext_vector_type(8)))  __bf16 v8bf;
typedef __attribute__((ext_vector_type(8)))  float  v8f;

union U16bf { v16bf v; v8bf h[2]; };

// ---------------------------------------------------------------------------
// prep1: per row h of W1: dot = W1[h,:]·x, r = ||W1[h,:]||^2
//        u0 = tanh(dot + b1), s = 1-u0^2, g = -2*u0*s*r
// ---------------------------------------------------------------------------
__global__ __launch_bounds__(256) void prep1(const float* __restrict__ x,
                                             const float* __restrict__ W1,
                                             const float* __restrict__ b1,
                                             float* __restrict__ u0v,
                                             float* __restrict__ sv,
                                             float* __restrict__ gv) {
  const int h = blockIdx.x;
  const float* row = W1 + (size_t)h * DC;
  float dot = 0.f, rr = 0.f;
  for (int i = threadIdx.x; i < DC; i += 256) {
    float w = row[i];
    dot += w * x[i];
    rr  += w * w;
  }
  for (int off = 16; off; off >>= 1) {
    dot += __shfl_down(dot, off);
    rr  += __shfl_down(rr,  off);
  }
  __shared__ float sd[8], sr[8];
  if ((threadIdx.x & 31) == 0) { sd[threadIdx.x >> 5] = dot; sr[threadIdx.x >> 5] = rr; }
  __syncthreads();
  if (threadIdx.x == 0) {
    float dsum = 0.f, rsum = 0.f;
    for (int q = 0; q < 8; ++q) { dsum += sd[q]; rsum += sr[q]; }
    float a0 = dsum + b1[h];
    float u  = tanhf(a0);
    float s  = 1.f - u * u;
    u0v[h] = u;
    sv[h]  = s;
    gv[h]  = -2.f * u * s * rsum;
  }
}

// ---------------------------------------------------------------------------
// transposeB: Bt[i, h] = bf16(W1[h, i]); 32x32 LDS tiles, both sides coalesced
// ---------------------------------------------------------------------------
__global__ __launch_bounds__(256) void transposeB(const float* __restrict__ W1,
                                                  __bf16* __restrict__ Bt) {
  __shared__ float tile[32][33];
  const int i0 = blockIdx.x * 32;   // D dimension
  const int h0 = blockIdx.y * 32;   // H1 dimension
  for (int r = threadIdx.y; r < 32; r += 8)
    tile[r][threadIdx.x] = W1[(size_t)(h0 + r) * DC + i0 + threadIdx.x];
  __syncthreads();
  for (int r = threadIdx.y; r < 32; r += 8)
    Bt[(size_t)(i0 + r) * H1C + h0 + threadIdx.x] = (__bf16)tile[threadIdx.x][r];
}

// ---------------------------------------------------------------------------
// prep2: per row j of W2: z0 = W2[j,:]·u0 + b2, S2 = W2[j,:]·g,
//        A[j,h] = bf16(W2[j,h]*s[h]) (fused with the single W2 read)
//        c1[j] = W3[j]*t*S2 ; c2[j] = 2*W3[j]*t*v0
// ---------------------------------------------------------------------------
__global__ __launch_bounds__(256) void prep2(const float* __restrict__ W2,
                                             const float* __restrict__ b2,
                                             const float* __restrict__ W3,
                                             const float* __restrict__ u0v,
                                             const float* __restrict__ sv,
                                             const float* __restrict__ gv,
                                             __bf16* __restrict__ Abf,
                                             float* __restrict__ c1,
                                             float* __restrict__ c2) {
  const int j = blockIdx.x;
  const float* row = W2 + (size_t)j * H1C;
  float z = 0.f, s2 = 0.f;
  for (int h = threadIdx.x; h < H1C; h += 256) {
    float w = row[h];
    z  += w * u0v[h];
    s2 += w * gv[h];
    Abf[(size_t)j * H1C + h] = (__bf16)(w * sv[h]);
  }
  for (int off = 16; off; off >>= 1) {
    z  += __shfl_down(z,  off);
    s2 += __shfl_down(s2, off);
  }
  __shared__ float sz[8], ss[8];
  if ((threadIdx.x & 31) == 0) { sz[threadIdx.x >> 5] = z; ss[threadIdx.x >> 5] = s2; }
  __syncthreads();
  if (threadIdx.x == 0) {
    float zs = 0.f, s2s = 0.f;
    for (int q = 0; q < 8; ++q) { zs += sz[q]; s2s += ss[q]; }
    float z0   = zs + b2[j];
    float v    = tanhf(z0);
    float t    = 1.f - v * v;
    float coef = W3[j] * t;
    c1[j] = coef * s2s;
    c2[j] = 2.f * coef * v;
  }
}

// ---------------------------------------------------------------------------
// gemm_s1: M = A(bf16) x Bt(bf16)^T with f32 accum via v_wmma_f32_16x16x32_bf16.
// Wave tile: 32 rows (j) x 64 cols (i); K loop over H1 in steps of 32.
// Afterwards: per-row sum of squares of the tile -> S1part[it * H2 + j]
// (deterministic partials, reduced in finalize; no float atomics).
// A 16x32 / B 32x16 use contiguous K (k = 16*half + s) in both operands; f32 C 16x16 in ISA layout.
// ---------------------------------------------------------------------------
__global__ __launch_bounds__(256) void gemm_s1(const __bf16* __restrict__ A,
                                               const __bf16* __restrict__ Bt,
                                               float* __restrict__ S1part) {
  const int lane = threadIdx.x & 31;
  const int wib  = threadIdx.x >> 5;
  const int w    = blockIdx.x * 8 + wib;   // 2048 waves total
  const int jt   = w >> 4;                 // 0..127 : 32-row block in H2
  const int it   = w & 15;                 // 0..15  : 64-col block in D
  const int jbase = jt * 32;
  const int ibase = it * 64;

  const int m  = lane & 15;        // row (A) / col (B) within fragment
  const int hh = lane >> 4;        // lane-half selects K sub-range

  const __bf16* arow0 = A + (size_t)(jbase + m) * H1C;
  const __bf16* arow1 = A + (size_t)(jbase + 16 + m) * H1C;
  const __bf16* brow[4];
#pragma unroll
  for (int c = 0; c < 4; ++c)
    brow[c] = Bt + (size_t)(ibase + c * 16 + m) * H1C;

  const int ak = hh * 16;   // A: halves own K {0..15} / {16..31}
  const int bk = hh * 16;   // B: halves own K {0..15} / {16..31}

  v8f acc[2][4];
#pragma unroll
  for (int r = 0; r < 2; ++r)
#pragma unroll
    for (int c = 0; c < 4; ++c)
      acc[r][c] = v8f{};

  for (int kk = 0; kk < H1C; kk += 32) {
    U16bf a0, a1;
    a0.h[0] = *(const v8bf*)(arow0 + kk + ak);
    a0.h[1] = *(const v8bf*)(arow0 + kk + ak + 8);
    a1.h[0] = *(const v8bf*)(arow1 + kk + ak);
    a1.h[1] = *(const v8bf*)(arow1 + kk + ak + 8);
    v16bf b[4];
#pragma unroll
    for (int c = 0; c < 4; ++c)
      b[c] = *(const v16bf*)(brow[c] + kk + bk);
#pragma unroll
    for (int c = 0; c < 4; ++c) {
      acc[0][c] = __builtin_amdgcn_wmma_f32_16x16x32_bf16(
          false, a0.v, false, b[c], (short)0, acc[0][c], false, false);
      acc[1][c] = __builtin_amdgcn_wmma_f32_16x16x32_bf16(
          false, a1.v, false, b[c], (short)0, acc[1][c], false, false);
    }
  }

  // C layout: element v, lanes 0-15 -> row v, lanes 16-31 -> row v+8 (cols = m).
  // Sum squares over the wave's 64 columns, then over the 16-lane group.
#pragma unroll
  for (int r = 0; r < 2; ++r) {
#pragma unroll
    for (int v = 0; v < 8; ++v) {
      float p = 0.f;
#pragma unroll
      for (int c = 0; c < 4; ++c) {
        float e = acc[r][c][v];
        p += e * e;
      }
      p += __shfl_xor(p, 1);
      p += __shfl_xor(p, 2);
      p += __shfl_xor(p, 4);
      p += __shfl_xor(p, 8);
      if (m == 0) {
        int jrow = jbase + r * 16 + v + hh * 8;
        S1part[(size_t)it * H2C + jrow] = p;
      }
    }
  }
}

// ---------------------------------------------------------------------------
// finalize: out = sum_j c1[j] - c2[j] * S1[j],  S1[j] = sum of 16 partials
// ---------------------------------------------------------------------------
__global__ __launch_bounds__(256) void finalize(const float* __restrict__ c1,
                                                const float* __restrict__ c2,
                                                const float* __restrict__ S1part,
                                                float* __restrict__ out) {
  float acc = 0.f;
  for (int j = threadIdx.x; j < H2C; j += 256) {
    float s1 = 0.f;
#pragma unroll
    for (int p = 0; p < 16; ++p) s1 += S1part[(size_t)p * H2C + j];
    acc += c1[j] - c2[j] * s1;
  }
  for (int off = 16; off; off >>= 1) acc += __shfl_down(acc, off);
  __shared__ float sm[8];
  if ((threadIdx.x & 31) == 0) sm[threadIdx.x >> 5] = acc;
  __syncthreads();
  if (threadIdx.x == 0) {
    float t = 0.f;
    for (int q = 0; q < 8; ++q) t += sm[q];
    out[0] = t;
  }
}

// ---------------------------------------------------------------------------
extern "C" void kernel_launch(void* const* d_in, const int* in_sizes, int n_in,
                              void* d_out, int out_size, void* d_ws, size_t ws_size,
                              hipStream_t stream) {
  const float* x  = (const float*)d_in[0];
  const float* W1 = (const float*)d_in[1];
  const float* b1 = (const float*)d_in[2];
  const float* W2 = (const float*)d_in[3];
  const float* b2 = (const float*)d_in[4];
  const float* W3 = (const float*)d_in[5];
  // d_in[6] (b3) does not appear in the Laplacian.
  float* out = (float*)d_out;

  // Workspace layout (~40.3 MB):
  __bf16* Abf  = (__bf16*)d_ws;                         // H2*H1 bf16 = 32 MB
  __bf16* Btbf = Abf + (size_t)H2C * H1C;               // D*H1  bf16 =  8 MB
  float*  u0v  = (float*)(Btbf + (size_t)DC * H1C);
  float*  sv   = u0v + H1C;
  float*  gv   = sv  + H1C;
  float*  c1   = gv  + H1C;
  float*  c2   = c1  + H2C;
  float*  S1p  = c2  + H2C;                             // 16*H2 floats = 256 KB

  prep1<<<H1C, 256, 0, stream>>>(x, W1, b1, u0v, sv, gv);
  transposeB<<<dim3(DC / 32, H1C / 32), dim3(32, 8), 0, stream>>>(W1, Btbf);
  prep2<<<H2C, 256, 0, stream>>>(W2, b2, W3, u0v, sv, gv, Abf, c1, c2);
  gemm_s1<<<256, 256, 0, stream>>>(Abf, Btbf, S1p);     // 2048 waves
  finalize<<<1, 256, 0, stream>>>(c1, c2, S1p, out);
}
